// gated_tpp_22024592294207
// MI455X (gfx1250) — matrix operand order repaired and checked by reading, not compiled, not run on hardware
//
#include <hip/hip_runtime.h>
#include <hip/hip_bf16.h>
#include <cmath>

typedef __attribute__((ext_vector_type(16))) _Float16 v16h;
typedef __attribute__((ext_vector_type(8)))  float    v8f;

#define D_MODEL 256
#define SEQ     4096
#define BATCH   4

// ---------------------------------------------------------------------------
// Prep 1: sinusoidal embedding, f16, stored TRANSPOSED as embT[b][d][j] so the
// WMMA B-tiles are contiguous 32B loads per lane.  8 MB -> lives in L2.
// ---------------------------------------------------------------------------
__global__ __launch_bounds__(256) void emb_prep_kernel(
    const float* __restrict__ et, _Float16* __restrict__ embT) {
  int idx  = blockIdx.x * 256 + threadIdx.x;      // over B*D*S
  int j    = idx & (SEQ - 1);
  int rest = idx >> 12;
  int d    = rest & (D_MODEL - 1);
  int b    = rest >> 8;
  float t    = et[b * SEQ + j];
  float e2   = 2.0f * (float)((d >> 1) + 1) * (1.0f / D_MODEL);
  float invw = __expf(-e2 * 9.210340371976184f);  // 10000^-e2
  float ph   = t * invw;
  float v    = (d & 1) ? __cosf(ph) : __sinf(ph);
  embT[idx] = (_Float16)v;
}

// Prep 2: W_in -> f16 (128 KB, L2-resident)
__global__ __launch_bounds__(256) void w_prep_kernel(
    const float* __restrict__ W, _Float16* __restrict__ w16) {
  int idx = blockIdx.x * 256 + threadIdx.x;       // over D*D
  w16[idx] = (_Float16)W[idx];
}

// ---------------------------------------------------------------------------
// Main fused kernel.  Block = 16 output rows, 4 waves (wave32).
// Each wave owns a 64-wide D slice = 4 WMMA tiles, so each score A-tile
// (16 exp/lane) is amortized over 4 v_wmma ops -> TRANS pipe no longer the
// bottleneck.  Score matrix is never materialized; B-tiles stream from the
// L2-resident transposed f16 embedding.
// ---------------------------------------------------------------------------
__global__ __launch_bounds__(128) void tpp_main_kernel(
    const float* __restrict__ et, const _Float16* __restrict__ embT,
    const _Float16* __restrict__ w16, const float* __restrict__ w_t,
    const float* __restrict__ b_t, const float* __restrict__ gamma,
    const float* __restrict__ beta, float* __restrict__ out) {

  __shared__ float    tAll[SEQ];              // 16 KB: all times of this batch
  __shared__ float    hid[16][D_MODEL];       // 16 KB: hidden (f32)
  __shared__ _Float16 hidh[16][D_MODEL];      //  8 KB: LN'd hidden (f16)
  __shared__ float    outacc[16];

  const int b    = blockIdx.y;
  const int i0   = blockIdx.x * 16;
  const int tid  = threadIdx.x;
  const int wave = tid >> 5;                  // 0..3
  const int lane = tid & 31;
  const int lo16 = lane & 15;
  const int hi   = lane >> 4;                 // 0 | 1 (half-wave)

  for (int k = tid; k < SEQ; k += 128) tAll[k] = et[b * SEQ + k];
  if (tid < 16) outacc[tid] = 0.0f;
  __syncthreads();

  const int   iRow = i0 + lo16;               // A-tile row per lane
  const float ti   = tAll[iRow];

  const int d0 = wave * 64;                   // this wave's 64-wide D slice
  v8f c[4] = {v8f{}, v8f{}, v8f{}, v8f{}};

  const _Float16* embB = embT + (size_t)b * D_MODEL * SEQ;
  const _Float16* pb[4];
  #pragma unroll
  for (int n = 0; n < 4; ++n)
    pb[n] = embB + (size_t)(d0 + n * 16 + lo16) * SEQ + hi * 16;

  const int kEnd = i0 + 16;                   // causal: only j <= i0+15
  for (int k0 = 0; k0 < kEnd; k0 += 32) {
    const bool full = (k0 + 31 <= i0);        // chunk fully below diagonal
    v16h a;
    #pragma unroll
    for (int e = 0; e < 16; ++e) {            // A layout: K = e + 16*hi
      int   K  = e + hi * 16;
      int   j  = k0 + K;
      float dt = ti - tAll[j];
      float s  = __expf(-0.5f * dt * dt);
      if (!full && j > iRow) s = 0.0f;        // causal mask on diagonal chunk
      a[e] = (_Float16)s;
    }
    v16h bt[4];
    #pragma unroll
    for (int n = 0; n < 4; ++n)               // 4 contiguous 32B B-tiles
      bt[n] = *(const v16h*)(pb[n] + k0);
    #pragma unroll
    for (int n = 0; n < 4; ++n)               // one A feeds 4 WMMAs
      c[n] = __builtin_amdgcn_wmma_f32_16x16x32_f16(false, a, false, bt[n],
                                                    (short)0, c[n], false, false);
  }

  // scatter C tiles: c[n][r] -> row (r + 8*hi), col d0 + n*16 + lo16
  #pragma unroll
  for (int n = 0; n < 4; ++n)
    #pragma unroll
    for (int r = 0; r < 8; ++r)
      hid[r + hi * 8][d0 + n * 16 + lo16] = c[n][r];
  __syncthreads();

  // LayerNorm: wave w handles rows 4w..4w+3 (shuffle reduction over D)
  #pragma unroll
  for (int rr = 0; rr < 4; ++rr) {
    int   row = wave * 4 + rr;
    float s = 0.f, q = 0.f;
    #pragma unroll
    for (int k = lane; k < D_MODEL; k += 32) {
      float v = hid[row][k]; s += v; q += v * v;
    }
    #pragma unroll
    for (int m = 16; m >= 1; m >>= 1) {
      s += __shfl_xor(s, m, 32);
      q += __shfl_xor(q, m, 32);
    }
    float mu  = s * (1.0f / D_MODEL);
    float var = q * (1.0f / D_MODEL) - mu * mu;
    float rs  = rsqrtf(var + 1e-6f);
    for (int k = lane; k < D_MODEL; k += 32) {
      float hn = (hid[row][k] - mu) * rs * gamma[k] + beta[k];
      hidh[row][k] = (_Float16)hn;
    }
  }
  __syncthreads();

  // Decoder GEMM: hs = relu(hidden @ W_in^T); wave owns 64 e-columns.
  v8f g[4] = {v8f{}, v8f{}, v8f{}, v8f{}};
  const int e0 = wave * 64;
  #pragma unroll
  for (int dk = 0; dk < D_MODEL; dk += 32) {
    v16h a2;
    #pragma unroll
    for (int e = 0; e < 16; ++e) {
      int K = e + hi * 16;
      a2[e] = hidh[lo16][dk + K];
    }
    #pragma unroll
    for (int n = 0; n < 4; ++n) {
      v16h bw = *(const v16h*)(w16 + (size_t)(e0 + n * 16 + lo16) * D_MODEL
                               + dk + hi * 16);
      g[n] = __builtin_amdgcn_wmma_f32_16x16x32_f16(false, a2, false, bw,
                                                    (short)0, g[n], false, false);
    }
  }

  // relu -> *w_t[e] -> reduce over e (lanes within half-wave, then LDS atomics)
  float wt[4];
  #pragma unroll
  for (int n = 0; n < 4; ++n) wt[n] = w_t[e0 + n * 16 + lo16];
  float p[8];
  #pragma unroll
  for (int r = 0; r < 8; ++r) {
    float acc = 0.f;
    #pragma unroll
    for (int n = 0; n < 4; ++n) {
      float h = g[n][r] > 0.f ? g[n][r] : 0.f;
      acc += h * wt[n];
    }
    p[r] = acc;
  }
  #pragma unroll
  for (int m = 8; m >= 1; m >>= 1) {
    #pragma unroll
    for (int r = 0; r < 8; ++r) p[r] += __shfl_xor(p[r], m, 32);
  }
  if (lo16 == 0) {
    #pragma unroll
    for (int r = 0; r < 8; ++r) atomicAdd(&outacc[r + hi * 8], p[r]);
  }
  __syncthreads();

  if (tid < 16) {
    float x  = outacc[tid] + b_t[0];
    float sp = (x > 20.f) ? x : log1pf(__expf(x));
    out[b * SEQ + i0 + tid] = sp;
  }
}

// ---------------------------------------------------------------------------
extern "C" void kernel_launch(void* const* d_in, const int* in_sizes, int n_in,
                              void* d_out, int out_size, void* d_ws, size_t ws_size,
                              hipStream_t stream) {
  (void)in_sizes; (void)n_in; (void)out_size; (void)ws_size;
  // inputs: 0 event_type(i32, unused) 1 event_time 2 arrival_times(unused)
  //         3 W_in 4 w_t 5 b_t 6 ln_gamma 7 ln_beta
  const float* event_time = (const float*)d_in[1];
  const float* W_in       = (const float*)d_in[3];
  const float* w_t        = (const float*)d_in[4];
  const float* b_t        = (const float*)d_in[5];
  const float* ln_gamma   = (const float*)d_in[6];
  const float* ln_beta    = (const float*)d_in[7];

  _Float16* embT = (_Float16*)d_ws;                       // 8 MB
  _Float16* w16  = embT + (size_t)BATCH * D_MODEL * SEQ;  // +128 KB

  emb_prep_kernel<<<(BATCH * D_MODEL * SEQ) / 256, 256, 0, stream>>>(event_time, embT);
  w_prep_kernel<<<(D_MODEL * D_MODEL) / 256, 256, 0, stream>>>(W_in, w16);

  dim3 grid(SEQ / 16, BATCH);
  tpp_main_kernel<<<grid, 128, 0, stream>>>(event_time, embT, w16, w_t, b_t,
                                            ln_gamma, ln_beta, (float*)d_out);
}
